// Colorizer_68616397521133
// MI455X (gfx1250) — matrix operand order repaired and checked by reading, not compiled, not run on hardware
//
#include <hip/hip_runtime.h>
#include <hip/hip_bf16.h>

typedef __attribute__((ext_vector_type(16))) _Float16 v16h;
typedef __attribute__((ext_vector_type(8)))  _Float16 h8;
typedef __attribute__((ext_vector_type(8)))  float    v8f;

// ---------------------------------------------------------------------------
// Kernel 1: convert f32 [b, C, HW] -> f16 [b, HW, C]  (transpose to K-major)
// ---------------------------------------------------------------------------
__global__ void k_cvt_transpose(const float* __restrict__ in, _Float16* __restrict__ out,
                                int C, int HW, int total) {
  int i = blockIdx.x * blockDim.x + threadIdx.x;
  if (i >= total) return;
  int c = i % C;
  int t = i / C;
  int p = t % HW;
  int b = t / HW;
  out[i] = (_Float16)in[((size_t)b * C + c) * (size_t)HW + p];
}

// ---------------------------------------------------------------------------
// Kernel 2: pairwise Gram GEMM  G[b,m,n] = sum_k A[b,m,k] * B[b,n,k]
// One wave32 per 16x16 output tile, v_wmma_f32_16x16x32_f16 over K.
// Lane layouts per CDNA5 ISA §7.12.2:
//   A (16x32 f16): lane L (half=L>>4, m=L&15): element e -> K = ((e<8)?e:e+8) + 8*half
//   B (32x16 f16): lane L (half=L>>4, n=L&15): element e -> K = ((e<8)?e:e+8) + 8*half
//   C/D (16x16 f32): VGPR r -> M = r + 8*half, N = L&15
// ---------------------------------------------------------------------------
__global__ __launch_bounds__(32) void k_wmma_gemm(const _Float16* __restrict__ A,
                                                  const _Float16* __restrict__ Bm,
                                                  float* __restrict__ G,
                                                  int M, int N, int K) {
  int nt_n = N >> 4, nt_m = M >> 4;
  int tile = blockIdx.x;
  int b  = tile / (nt_m * nt_n);
  int r  = tile % (nt_m * nt_n);
  int mt = r / nt_n, nt = r % nt_n;
  int lane = threadIdx.x & 31;
  int half = lane >> 4, lm = lane & 15;
  const _Float16* Arow = A  + ((size_t)b * M + (size_t)mt * 16 + lm) * (size_t)K;
  const _Float16* Brow = Bm + ((size_t)b * N + (size_t)nt * 16 + lm) * (size_t)K;
  v8f acc = {};
  for (int kb = 0; kb < K; kb += 32) {
    h8 alo = *(const h8*)(Arow + kb + 8 * half);        // K = kb + 8*half + 0..7
    h8 ahi = *(const h8*)(Arow + kb + 16 + 8 * half);   // K = kb + 16 + 8*half + 0..7
    const h8* bp = (const h8*)(Brow + kb + 8 * half);   // K = kb + 8*half + 0..7, kb + 16 + 8*half + 0..7
    h8 b0 = bp[0], b1 = bp[2];
    v16h a, bb;
    #pragma unroll
    for (int e = 0; e < 8; ++e) {
      a[e] = alo[e];  a[8 + e] = ahi[e];
      bb[e] = b0[e];  bb[8 + e] = b1[e];
    }
    acc = __builtin_amdgcn_wmma_f32_16x16x32_f16(false, a, false, bb, (short)0, acc,
                                                 false, false);
  }
  float* Gb = G + (size_t)b * M * N;
  #pragma unroll
  for (int rr = 0; rr < 8; ++rr) {
    int m = mt * 16 + rr + 8 * half;
    int n = nt * 16 + lm;
    Gb[(size_t)m * N + n] = acc[rr];
  }
}

// ---------------------------------------------------------------------------
// Kernel 3: coarse dilated 13x13 lookup from G0 -> softmax -> soft-argmax
// offsets.  One thread per pixel.
// ---------------------------------------------------------------------------
__global__ void k_coarse_offset(const float* __restrict__ G0, float* __restrict__ off, int B) {
  int idx = blockIdx.x * blockDim.x + threadIdx.x;
  if (idx >= B * 1024) return;
  int y = (idx & 1023) >> 5, x = idx & 31;
  const float* g = G0 + (size_t)idx * 1024;
  float mx = -1e30f;
  for (int i = 0; i < 13; ++i)
    for (int j = 0; j < 13; ++j) {
      int qy = y + (i - 6) * 2, qx = x + (j - 6) * 2;
      float v = ((unsigned)qy < 32u && (unsigned)qx < 32u) ? g[qy * 32 + qx] : 0.0f;
      mx = fmaxf(mx, v);
    }
  float se = 0.0f, sx = 0.0f, sy = 0.0f;
  for (int i = 0; i < 13; ++i)
    for (int j = 0; j < 13; ++j) {
      int qy = y + (i - 6) * 2, qx = x + (j - 6) * 2;
      float v = ((unsigned)qy < 32u && (unsigned)qx < 32u) ? g[qy * 32 + qx] : 0.0f;
      float e = __expf(v - mx);
      se += e;
      sx += e * (float)(j - 6);
      sy += e * (float)(i - 6);
    }
  float inv = 2.0f / se;  // * DIRATE
  off[2 * idx + 0] = sx * inv;   // off_x
  off[2 * idx + 1] = sy * inv;   // off_y
}

__device__ __forceinline__ float wred_max(float v) {
  #pragma unroll
  for (int o = 16; o > 0; o >>= 1) v = fmaxf(v, __shfl_xor(v, o, 32));
  return v;
}
__device__ __forceinline__ float wred_sum(float v) {
  #pragma unroll
  for (int o = 16; o > 0; o >>= 1) v += __shfl_xor(v, o, 32);
  return v;
}

// ---------------------------------------------------------------------------
// Kernel 4: per pixel — corr0 (bilinear lookup of G0), corr1 (lookup of G1),
// joint softmax over 338, weighted sum of deform/unfold color patches.
// One wave32 per pixel, 338 logits split across lanes (11 register slots).
// ---------------------------------------------------------------------------
__global__ void k_pixel_out(const float* __restrict__ G0, const float* __restrict__ G1,
                            const float* __restrict__ off,
                            const float* __restrict__ q0, const float* __restrict__ q1,
                            float* __restrict__ out, int B) {
  int wave = threadIdx.x >> 5;
  int lane = threadIdx.x & 31;
  int idx = blockIdx.x * (blockDim.x >> 5) + wave;
  if (idx >= B * 1024) return;
  int b = idx >> 10, p = idx & 1023;
  int y = p >> 5, x = p & 31;
  const float* g0 = G0 + (size_t)idx * 1024;
  const float* g1 = G1 + (size_t)idx * 1024;
  float ox = off[2 * idx + 0], oy = off[2 * idx + 1];

  float lg[11];
  float mx = -1e30f;
  #pragma unroll
  for (int s = 0; s < 11; ++s) {
    int n = lane + 32 * s;
    float v = 0.0f;
    if (n < 338) {
      if (n < 169) {                     // deformable corr0 via bilinear G0 lookup
        int ky = n / 13 - 6, kx = n % 13 - 6;
        float sy = oy + (float)(y + ky);
        float sx = ox + (float)(x + kx);
        float fy = floorf(sy), fx = floorf(sx);
        int iy = (int)fy, ix = (int)fx;
        float wy = sy - fy, wx = sx - fx;
        float v00 = ((unsigned)iy       < 32u && (unsigned)ix       < 32u) ? g0[iy * 32 + ix]           : 0.0f;
        float v01 = ((unsigned)iy       < 32u && (unsigned)(ix + 1) < 32u) ? g0[iy * 32 + ix + 1]       : 0.0f;
        float v10 = ((unsigned)(iy + 1) < 32u && (unsigned)ix       < 32u) ? g0[(iy + 1) * 32 + ix]     : 0.0f;
        float v11 = ((unsigned)(iy + 1) < 32u && (unsigned)(ix + 1) < 32u) ? g0[(iy + 1) * 32 + ix + 1] : 0.0f;
        v = v00 * (1.0f - wy) * (1.0f - wx) + v01 * (1.0f - wy) * wx
          + v10 * wy * (1.0f - wx)          + v11 * wy * wx;
      } else {                           // corr1: integer lookup of G1
        int n2 = n - 169;
        int dy = n2 / 13 - 6, dx = n2 % 13 - 6;
        int qy = y + dy, qx = x + dx;
        v = ((unsigned)qy < 32u && (unsigned)qx < 32u) ? g1[qy * 32 + qx] : 0.0f;
      }
      mx = fmaxf(mx, v);
    }
    lg[s] = v;
  }
  mx = wred_max(mx);

  float se = 0.0f;
  #pragma unroll
  for (int s = 0; s < 11; ++s) {
    int n = lane + 32 * s;
    if (n < 338) { float e = __expf(lg[s] - mx); lg[s] = e; se += e; }
  }
  se = wred_sum(se);

  float a0 = 0.0f, a1 = 0.0f, a2 = 0.0f;
  #pragma unroll
  for (int s = 0; s < 11; ++s) {
    int n = lane + 32 * s;
    if (n >= 338) continue;
    float e = lg[s];
    if (n < 169) {                       // uf0: bilinear sample of qr0 (::4,::4)
      int ky = n / 13 - 6, kx = n % 13 - 6;
      float sy = oy + (float)(y + ky);
      float sx = ox + (float)(x + kx);
      float fy = floorf(sy), fx = floorf(sx);
      int iy = (int)fy, ix = (int)fx;
      float wy = sy - fy, wx = sx - fx;
      float w00 = (1.0f - wy) * (1.0f - wx), w01 = (1.0f - wy) * wx;
      float w10 = wy * (1.0f - wx),          w11 = wy * wx;
      bool m00 = (unsigned)iy       < 32u && (unsigned)ix       < 32u;
      bool m01 = (unsigned)iy       < 32u && (unsigned)(ix + 1) < 32u;
      bool m10 = (unsigned)(iy + 1) < 32u && (unsigned)ix       < 32u;
      bool m11 = (unsigned)(iy + 1) < 32u && (unsigned)(ix + 1) < 32u;
      #pragma unroll
      for (int ch = 0; ch < 3; ++ch) {
        const float* plane = q0 + ((size_t)(b * 3 + ch)) * 16384;  // 128x128, stride-4 sampled
        float u = 0.0f;
        if (m00) u += w00 * plane[iy * 512 + ix * 4];
        if (m01) u += w01 * plane[iy * 512 + (ix + 1) * 4];
        if (m10) u += w10 * plane[(iy + 1) * 512 + ix * 4];
        if (m11) u += w11 * plane[(iy + 1) * 512 + (ix + 1) * 4];
        float ew = e * u;
        if (ch == 0) a0 += ew; else if (ch == 1) a1 += ew; else a2 += ew;
      }
    } else {                             // uf1: unfold of qr1 (::4,::4)
      int n2 = n - 169;
      int dy = n2 / 13 - 6, dx = n2 % 13 - 6;
      int qy = y + dy, qx = x + dx;
      if ((unsigned)qy < 32u && (unsigned)qx < 32u) {
        #pragma unroll
        for (int ch = 0; ch < 3; ++ch) {
          const float* plane = q1 + ((size_t)(b * 3 + ch)) * 16384;
          float ew = e * plane[qy * 512 + qx * 4];
          if (ch == 0) a0 += ew; else if (ch == 1) a1 += ew; else a2 += ew;
        }
      }
    }
  }
  a0 = wred_sum(a0); a1 = wred_sum(a1); a2 = wred_sum(a2);
  if (lane == 0) {
    float inv = 1.0f / se;
    out[(size_t)(b * 3 + 0) * 1024 + p] = a0 * inv;
    out[(size_t)(b * 3 + 1) * 1024 + p] = a1 * inv;
    out[(size_t)(b * 3 + 2) * 1024 + p] = a2 * inv;
  }
}

// ---------------------------------------------------------------------------
// Kernel 5: semantic region 7x7 lookup from Gsem -> softmax -> second output.
// ---------------------------------------------------------------------------
__global__ void k_region(const float* __restrict__ Gs, float* __restrict__ outc, int B) {
  int idx = blockIdx.x * blockDim.x + threadIdx.x;
  if (idx >= B * 256) return;
  int b = idx >> 8, p = idx & 255;
  int y = p >> 4, x = p & 15;
  const float* g = Gs + (size_t)idx * 256;
  float mx = -1e30f;
  for (int n = 0; n < 49; ++n) {
    int qy = y + n / 7 - 3, qx = x + n % 7 - 3;
    float v = ((unsigned)qy < 16u && (unsigned)qx < 16u) ? g[qy * 16 + qx] : 0.0f;
    mx = fmaxf(mx, v);
  }
  float se = 0.0f;
  for (int n = 0; n < 49; ++n) {
    int qy = y + n / 7 - 3, qx = x + n % 7 - 3;
    float v = ((unsigned)qy < 16u && (unsigned)qx < 16u) ? g[qy * 16 + qx] : 0.0f;
    se += __expf(v - mx);
  }
  float inv = 1.0f / se;
  for (int n = 0; n < 49; ++n) {
    int qy = y + n / 7 - 3, qx = x + n % 7 - 3;
    float v = ((unsigned)qy < 16u && (unsigned)qx < 16u) ? g[qy * 16 + qx] : 0.0f;
    outc[((size_t)b * 49 + n) * 256 + p] = __expf(v - mx) * inv;
  }
}

// ---------------------------------------------------------------------------
extern "C" void kernel_launch(void* const* d_in, const int* in_sizes, int n_in,
                              void* d_out, int out_size, void* d_ws, size_t ws_size,
                              hipStream_t stream) {
  (void)n_in; (void)out_size; (void)ws_size;
  const float* feats_r0 = (const float*)d_in[0];
  const float* feats_r1 = (const float*)d_in[1];
  const float* feats_t  = (const float*)d_in[2];
  const float* q_r0     = (const float*)d_in[3];
  const float* q_r1     = (const float*)d_in[4];
  const float* sem_r1   = (const float*)d_in[6];   // reference uses feats_r_semantic1
  const float* sem_t    = (const float*)d_in[7];
  int B = in_sizes[2] / (128 * 32 * 32);

  // Workspace carve (256B-aligned), ~19.5 MB total for B=2 — L2 resident.
  char* ws = (char*)d_ws;
  size_t o = 0;
  auto carve = [&](size_t bytes) -> void* {
    void* p = (void*)(ws + o);
    o += (bytes + 255) & ~(size_t)255;
    return p;
  };
  _Float16* Tt  = (_Float16*)carve((size_t)B * 1024 * 128 * 2);
  _Float16* R0t = (_Float16*)carve((size_t)B * 1024 * 128 * 2);
  _Float16* R1t = (_Float16*)carve((size_t)B * 1024 * 128 * 2);
  _Float16* Tst = (_Float16*)carve((size_t)B * 256 * 512 * 2);
  _Float16* Rst = (_Float16*)carve((size_t)B * 256 * 512 * 2);
  float* G0   = (float*)carve((size_t)B * 1024 * 1024 * 4);
  float* G1   = (float*)carve((size_t)B * 1024 * 1024 * 4);
  float* Gs   = (float*)carve((size_t)B * 256 * 256 * 4);
  float* offv = (float*)carve((size_t)B * 1024 * 2 * 4);

  // 1) f32 -> f16, transpose to [b, pixel, channel]
  int tp = B * 128 * 1024;
  k_cvt_transpose<<<(tp + 255) / 256, 256, 0, stream>>>(feats_t,  Tt,  128, 1024, tp);
  k_cvt_transpose<<<(tp + 255) / 256, 256, 0, stream>>>(feats_r0, R0t, 128, 1024, tp);
  k_cvt_transpose<<<(tp + 255) / 256, 256, 0, stream>>>(feats_r1, R1t, 128, 1024, tp);
  int ts = B * 512 * 256;
  k_cvt_transpose<<<(ts + 255) / 256, 256, 0, stream>>>(sem_t,  Tst, 512, 256, ts);
  k_cvt_transpose<<<(ts + 255) / 256, 256, 0, stream>>>(sem_r1, Rst, 512, 256, ts);

  // 2) pairwise Gram GEMMs via WMMA
  k_wmma_gemm<<<B * 64 * 64, 32, 0, stream>>>(Tt,  R0t, G0, 1024, 1024, 128);
  k_wmma_gemm<<<B * 64 * 64, 32, 0, stream>>>(Tt,  R1t, G1, 1024, 1024, 128);
  k_wmma_gemm<<<B * 16 * 16, 32, 0, stream>>>(Tst, Rst, Gs, 256, 256, 512);

  // 3) coarse softmax -> soft-argmax offsets
  k_coarse_offset<<<(B * 1024 + 127) / 128, 128, 0, stream>>>(G0, offv, B);

  // 4) joint 338-way softmax + color recombination -> out[b,3,32,32]
  k_pixel_out<<<(B * 1024 + 7) / 8, 256, 0, stream>>>(G0, G1, offv, q_r0, q_r1,
                                                      (float*)d_out, B);

  // 5) semantic 49-way softmax -> corr[b,1,49,256] (second tuple output)
  k_region<<<(B * 256 + 127) / 128, 128, 0, stream>>>(Gs, (float*)d_out + (size_t)B * 3072, B);
}
